// RouteNetModel_180388627358
// MI455X (gfx1250) — matrix operand order repaired and checked by reading, not compiled, not run on hardware
//
#include <hip/hip_runtime.h>
#include <hip/hip_bf16.h>

typedef __attribute__((ext_vector_type(16))) __bf16 v16bf;
typedef __attribute__((ext_vector_type(8)))  float  v8f;

union Frag16 { v16bf v; uint4 q[2]; };

__device__ __forceinline__ v8f wmma_bf16(v16bf a, v16bf b, v8f c) {
    // D = A(16x32 bf16) * B(32x16 bf16) + C(16x16 f32)
    return __builtin_amdgcn_wmma_f32_16x16x32_bf16(false, a, false, b, (short)0, c, false, false);
}

// A-fragment (16x32 bf16) from row-major [16][K] bf16 tile in LDS.
// lanes 0-15: row=lane, K={kt*32+0..15}
// lanes 16-31: row=lane-16, K={kt*32+16..31}
__device__ __forceinline__ v16bf ldsA(const __bf16* base, int row, int kt, int hi, int rowStrideU4) {
    const uint4* u = (const uint4*)base;
    int i0 = row * rowStrideU4 + kt * 4 + 2 * hi;
    Frag16 f;
    f.q[0] = u[i0];
    f.q[1] = u[i0 + 1];
    return f.v;
}

// B-fragment (32x16 bf16) from n-major [N][K] bf16 tile in LDS (K contiguous).
__device__ __forceinline__ v16bf ldsB(const __bf16* base, int nt, int kt, int lane, int kStrideU4) {
    int n = nt * 16 + (lane & 15);
    int hi = lane >> 4;
    const uint4* u = (const uint4*)base;
    int i0 = n * kStrideU4 + kt * 4 + (hi ? 2 : 0);
    Frag16 f;
    f.q[0] = u[i0];
    f.q[1] = u[i0 + 1];
    return f.v;
}

// Same but from global memory (bf16, n-major, K contiguous).
__device__ __forceinline__ v16bf glbB(const uint4* base, int nt, int kt, int lane, int kStrideU4) {
    int n = nt * 16 + (lane & 15);
    int hi = lane >> 4;
    long i0 = (long)n * kStrideU4 + kt * 4 + (hi ? 2 : 0);
    Frag16 f;
    f.q[0] = base[i0];
    f.q[1] = base[i0 + 1];
    return f.v;
}

__device__ __forceinline__ uint4 pack8(float4 a, float4 b) {
    union { uint4 u; __bf16 h[8]; } pk;
    pk.h[0] = (__bf16)a.x; pk.h[1] = (__bf16)a.y; pk.h[2] = (__bf16)a.z; pk.h[3] = (__bf16)a.w;
    pk.h[4] = (__bf16)b.x; pk.h[5] = (__bf16)b.y; pk.h[6] = (__bf16)b.z; pk.h[7] = (__bf16)b.w;
    return pk.u;
}

// fast sigmoid: v_exp_f32 + v_add + v_rcp_f32 (no IEEE div expansion)
__device__ __forceinline__ float sigm(float x) {
    return __builtin_amdgcn_rcpf(1.f + __expf(-x));
}
// fast tanh: CDNA5 V_TANH_F32 if the builtin exists, else rcp-based identity
__device__ __forceinline__ float tanh_f(float x) {
#if __has_builtin(__builtin_amdgcn_tanhf)
    return __builtin_amdgcn_tanhf(x);
#else
    float s = __builtin_amdgcn_rcpf(1.f + __expf(-2.f * x));
    return fmaf(2.f, s, -1.f);
#endif
}
__device__ __forceinline__ float selu_f(float x) {
    const float sc = 1.0507009873554805f, al = 1.6732632423543772f;
    return x > 0.f ? sc * x : sc * al * (__expf(x) - 1.f);
}

// ---------------------------------------------------------------- init kernels

__global__ void k_init_links(const float* __restrict__ lf, __bf16* __restrict__ ls, int n_links) {
    int idx = blockIdx.x * blockDim.x + threadIdx.x;
    if (idx >= n_links * 64) return;
    int l = idx >> 6, c = idx & 63;
    ls[idx] = (__bf16)((c < 7) ? lf[l * 7 + c] : 0.f);
}

__global__ void k_init_paths(const float* __restrict__ pf, float* __restrict__ s1,
                             float* __restrict__ s2, int n_paths) {
    int idx = blockIdx.x * blockDim.x + threadIdx.x;
    if (idx >= n_paths * 64) return;
    int p = idx >> 6, c = idx & 63;
    float v = (c < 11) ? pf[p * 11 + c] : 0.f;
    s1[idx] = v;
    s2[idx] = v;
}

__global__ void k_rowinfo(const int* __restrict__ paths, const int* __restrict__ seqs,
                          int* __restrict__ lens, int* __restrict__ rowst, int E) {
    int e = blockIdx.x * blockDim.x + threadIdx.x;
    if (e >= E) return;
    int p = paths[e], s = seqs[e];
    atomicMax(&lens[p], s + 1);
    if (s == 0) rowst[p] = e;
}

__global__ void k_prep_ro(const float* __restrict__ w1, const float* __restrict__ w2,
                          __bf16* __restrict__ w1t, __bf16* __restrict__ w2t) {
    int idx = blockIdx.x * blockDim.x + threadIdx.x;
    if (idx < 256 * 128) {
        int n = idx >> 7, k = idx & 127;          // w1: [128,256] -> w1t[n][k]
        w1t[n * 128 + k] = (__bf16)w1[k * 256 + n];
    } else if (idx < 256 * 128 + 256 * 256) {
        int j = idx - 256 * 128;
        int n = j >> 8, k = j & 255;              // w2: [256,256] -> w2t[n][k]
        w2t[n * 256 + k] = (__bf16)w2[k * 256 + n];
    }
}

// ---------------------------------------------------------------- GRU scan (WMMA)

__global__ __launch_bounds__(128) __attribute__((amdgpu_waves_per_eu(4)))
void k_gru(
    const int* __restrict__ links, const int* __restrict__ rowst, const int* __restrict__ lens,
    const __bf16* __restrict__ ls, float* __restrict__ agg,
    float* __restrict__ s1g, float* __restrict__ s2g,
    const float* __restrict__ g1k, const float* __restrict__ g1rk, const float* __restrict__ g1b,
    const float* __restrict__ g2k, const float* __restrict__ g2rk, const float* __restrict__ g2b,
    int n_paths)
{
    // bf16 transposed weights [192][64], 24 KB each (CDNA5: 320 KB LDS / WGP).
    __shared__ __attribute__((aligned(16))) __bf16 wK1[192 * 64];
    __shared__ __attribute__((aligned(16))) __bf16 wR1[192 * 64];
    __shared__ __attribute__((aligned(16))) __bf16 wK2[192 * 64];
    __shared__ __attribute__((aligned(16))) __bf16 wR2[192 * 64];
    __shared__ __attribute__((aligned(16))) __bf16 xStage[4][16 * 64];   // x tile, then n1 stage
    __shared__ __attribute__((aligned(16))) __bf16 s1St[4][16 * 64];     // persistent bf16 s1
    __shared__ __attribute__((aligned(16))) __bf16 s2St[4][16 * 64];     // persistent bf16 s2
    __shared__ float bL1[384], bL2[384];

    const int tid = threadIdx.x;
    for (int i = tid; i < 192 * 64; i += 128) {
        int n = i >> 6, k = i & 63;
        wK1[n * 64 + k] = (__bf16)g1k [k * 192 + n];
        wR1[n * 64 + k] = (__bf16)g1rk[k * 192 + n];
        wK2[n * 64 + k] = (__bf16)g2k [k * 192 + n];
        wR2[n * 64 + k] = (__bf16)g2rk[k * 192 + n];
    }
    for (int i = tid; i < 384; i += 128) { bL1[i] = g1b[i]; bL2[i] = g2b[i]; }
    __syncthreads();

    const int wave = tid >> 5, lane = tid & 31;
    const int hi = lane >> 4, hi8 = hi * 8, col = lane & 15;
    const int pbase = blockIdx.x * 64 + wave * 16;
    __bf16* xs  = xStage[wave];
    __bf16* s1s = s1St[wave];
    __bf16* s2s = s2St[wave];

    const v8f kZero = (v8f)(0.f);

    // pre-combined per-lane bias scalars (keras reset_after GRU: b[0] + b[1] for z,r;
    // b[0] and b[1] separate for the candidate because of the r-gating)
    float bz1[4], br1[4], bx1[4], bh1[4], bz2[4], br2[4], bx2[4], bh2[4];
#pragma unroll
    for (int j = 0; j < 4; ++j) {
        int c = j * 16 + col;
        bz1[j] = bL1[c]       + bL1[192 + c];
        br1[j] = bL1[64 + c]  + bL1[256 + c];
        bx1[j] = bL1[128 + c];
        bh1[j] = bL1[320 + c];
        bz2[j] = bL2[c]       + bL2[192 + c];
        br2[j] = bL2[64 + c]  + bL2[256 + c];
        bx2[j] = bL2[128 + c];
        bh2[j] = bL2[320 + c];
    }

    // persistent f32 state fragments (C/D layout: elem v -> row v+hi8, col 16*j+col)
    v8f s1f[4], s2f[4];
    int len8[8];
#pragma unroll
    for (int v = 0; v < 8; ++v) {
        int p = pbase + v + hi8;
        len8[v] = (p < n_paths) ? lens[p] : 0;
    }
#pragma unroll
    for (int j = 0; j < 4; ++j) {
#pragma unroll
        for (int v = 0; v < 8; ++v) {
            int p = pbase + v + hi8;
            float a = 0.f, b = 0.f;
            if (p < n_paths) {
                a = s1g[p * 64 + j * 16 + col];
                b = s2g[p * 64 + j * 16 + col];
            }
            s1f[j][v] = a;
            s2f[j][v] = b;
            s1s[(v + hi8) * 64 + j * 16 + col] = (__bf16)a;
            s2s[(v + hi8) * 64 + j * 16 + col] = (__bf16)b;
        }
    }
    // per-row CSR info (lanes 0..15, row = col)
    int rs = 0, lrow = 0;
    {
        int p = pbase + col;
        if (hi == 0 && p < n_paths) { rs = rowst[p]; lrow = lens[p]; }
    }

    for (int t = 0; t < 16; ++t) {
        // ---- link id per row via cross-lane shuffle (wave32)
        int lidval = -1;
        if (hi == 0) lidval = (t < lrow) ? links[rs + t] : -1;
        int lid8[8];
#pragma unroll
        for (int v = 0; v < 8; ++v) lid8[v] = __shfl(lidval, hi8 + v, 32);

        // ---- gather x tile [16][64] bf16 (link_state already bf16)
        {
            int row = lane >> 1, ch = lane & 1;
            int lid = __shfl(lidval, row, 32);
            const uint4* src = (const uint4*)(ls + (size_t)(lid < 0 ? 0 : lid) * 64 + ch * 32);
            uint4* dst = (uint4*)(xs + row * 64 + ch * 32);
            uint4 z4 = make_uint4(0u, 0u, 0u, 0u);
#pragma unroll
            for (int q = 0; q < 4; ++q)
                dst[q] = (lid >= 0) ? src[q] : z4;
        }
        __builtin_amdgcn_wave_barrier();
        v16bf ax[2], ah[2];
        ax[0] = ldsA(xs,  col, 0, hi, 8);
        ax[1] = ldsA(xs,  col, 1, hi, 8);
        ah[0] = ldsA(s1s, col, 0, hi, 8);
        ah[1] = ldsA(s1s, col, 1, hi, 8);
        __builtin_amdgcn_sched_barrier(0);

        // ---- GRU1: n1 = GRU(x, s1); commit s1 (mask known now), stage bf16(n1)
#pragma unroll
        for (int j = 0; j < 4; ++j) {
            v8f xz = wmma_bf16(ax[0], ldsB(wK1, j,     0, lane, 8), kZero);
            v8f xr = wmma_bf16(ax[0], ldsB(wK1, 4 + j, 0, lane, 8), kZero);
            v8f xh = wmma_bf16(ax[0], ldsB(wK1, 8 + j, 0, lane, 8), kZero);
            v8f hz = wmma_bf16(ah[0], ldsB(wR1, j,     0, lane, 8), kZero);
            v8f hr = wmma_bf16(ah[0], ldsB(wR1, 4 + j, 0, lane, 8), kZero);
            v8f hh = wmma_bf16(ah[0], ldsB(wR1, 8 + j, 0, lane, 8), kZero);
            xz = wmma_bf16(ax[1], ldsB(wK1, j,     1, lane, 8), xz);
            xr = wmma_bf16(ax[1], ldsB(wK1, 4 + j, 1, lane, 8), xr);
            xh = wmma_bf16(ax[1], ldsB(wK1, 8 + j, 1, lane, 8), xh);
            hz = wmma_bf16(ah[1], ldsB(wR1, j,     1, lane, 8), hz);
            hr = wmma_bf16(ah[1], ldsB(wR1, 4 + j, 1, lane, 8), hr);
            hh = wmma_bf16(ah[1], ldsB(wR1, 8 + j, 1, lane, 8), hh);
#pragma unroll
            for (int v = 0; v < 8; ++v) {
                float z = sigm(xz[v] + hz[v] + bz1[j]);
                float r = sigm(xr[v] + hr[v] + br1[j]);
                float c = tanh_f(xh[v] + bx1[j] + r * (hh[v] + bh1[j]));
                float n1 = z * s1f[j][v] + (1.f - z) * c;
                xs[(v + hi8) * 64 + j * 16 + col] = (__bf16)n1;     // n1 stage
                if (t < len8[v]) s1f[j][v] = n1;
                s1s[(v + hi8) * 64 + j * 16 + col] = (__bf16)s1f[j][v];
            }
            __builtin_amdgcn_sched_barrier(0);
        }

        __builtin_amdgcn_wave_barrier();
        v16bf an[2], a2[2];
        an[0] = ldsA(xs,  col, 0, hi, 8);
        an[1] = ldsA(xs,  col, 1, hi, 8);
        a2[0] = ldsA(s2s, col, 0, hi, 8);
        a2[1] = ldsA(s2s, col, 1, hi, 8);
        __builtin_amdgcn_sched_barrier(0);

        // ---- GRU2: n2 = GRU(n1, s2); masked commit + message scatter
#pragma unroll
        for (int j = 0; j < 4; ++j) {
            v8f xz = wmma_bf16(an[0], ldsB(wK2, j,     0, lane, 8), kZero);
            v8f xr = wmma_bf16(an[0], ldsB(wK2, 4 + j, 0, lane, 8), kZero);
            v8f xh = wmma_bf16(an[0], ldsB(wK2, 8 + j, 0, lane, 8), kZero);
            v8f hz = wmma_bf16(a2[0], ldsB(wR2, j,     0, lane, 8), kZero);
            v8f hr = wmma_bf16(a2[0], ldsB(wR2, 4 + j, 0, lane, 8), kZero);
            v8f hh = wmma_bf16(a2[0], ldsB(wR2, 8 + j, 0, lane, 8), kZero);
            xz = wmma_bf16(an[1], ldsB(wK2, j,     1, lane, 8), xz);
            xr = wmma_bf16(an[1], ldsB(wK2, 4 + j, 1, lane, 8), xr);
            xh = wmma_bf16(an[1], ldsB(wK2, 8 + j, 1, lane, 8), xh);
            hz = wmma_bf16(a2[1], ldsB(wR2, j,     1, lane, 8), hz);
            hr = wmma_bf16(a2[1], ldsB(wR2, 4 + j, 1, lane, 8), hr);
            hh = wmma_bf16(a2[1], ldsB(wR2, 8 + j, 1, lane, 8), hh);
#pragma unroll
            for (int v = 0; v < 8; ++v) {
                float z = sigm(xz[v] + hz[v] + bz2[j]);
                float r = sigm(xr[v] + hr[v] + br2[j]);
                float c = tanh_f(xh[v] + bx2[j] + r * (hh[v] + bh2[j]));
                float n2 = z * s2f[j][v] + (1.f - z) * c;
                if (t < len8[v]) {
                    s2f[j][v] = n2;
                    int lid = lid8[v];
                    if (lid >= 0)
                        atomicAdd(&agg[lid * 64 + j * 16 + col], n2);   // segment_sum
                }
                s2s[(v + hi8) * 64 + j * 16 + col] = (__bf16)s2f[j][v];
            }
            __builtin_amdgcn_sched_barrier(0);
        }
    }

    // write back persistent f32 states
#pragma unroll
    for (int j = 0; j < 4; ++j)
#pragma unroll
        for (int v = 0; v < 8; ++v) {
            int p = pbase + v + hi8;
            if (p < n_paths) {
                s1g[p * 64 + j * 16 + col] = s1f[j][v];
                s2g[p * 64 + j * 16 + col] = s2f[j][v];
            }
        }
}

// ---------------------------------------------------------------- link update MLP (tiny)

__global__ __launch_bounds__(128) void k_linkupd(
    const float* __restrict__ agg, __bf16* __restrict__ ls,
    const float* __restrict__ w1, const float* __restrict__ b1,
    const float* __restrict__ w2, const float* __restrict__ b2,
    const float* __restrict__ w3, const float* __restrict__ b3)
{
    __shared__ float sin_[128], sh1[128], sh2[128];
    int l = blockIdx.x, tid = threadIdx.x;
    sin_[tid] = (tid < 64) ? agg[l * 64 + tid] : (float)ls[l * 64 + tid - 64];
    __syncthreads();
    float a = b1[tid];
    for (int i = 0; i < 128; ++i) a += sin_[i] * w1[i * 128 + tid];
    sh1[tid] = selu_f(a);
    __syncthreads();
    a = b2[tid];
    for (int i = 0; i < 128; ++i) a += sh1[i] * w2[i * 128 + tid];
    sh2[tid] = fmaxf(a, 0.f);
    __syncthreads();
    if (tid < 64) {
        a = b3[tid];
        for (int i = 0; i < 128; ++i) a += sh2[i] * w3[i * 64 + tid];
        ls[l * 64 + tid] = (__bf16)a;
    }
}

// ---------------------------------------------------------------- readout (WMMA)

__global__ __launch_bounds__(128) __attribute__((amdgpu_waves_per_eu(4)))
void k_readout(
    const float* __restrict__ s1g, const float* __restrict__ s2g,
    const uint4* __restrict__ w1t, const uint4* __restrict__ w2t,
    const float* __restrict__ rb1, const float* __restrict__ rb2,
    const float* __restrict__ rw3, const float* __restrict__ rb3,
    const float* __restrict__ finw, const float* __restrict__ finb,
    float* __restrict__ out, int n_paths)
{
    __shared__ __attribute__((aligned(16))) __bf16 psS[4][16 * 128];
    __shared__ __attribute__((aligned(16))) __bf16 r1S[4][16 * 256];
    __shared__ float r3S[4][16];

    const int tid = threadIdx.x, wave = tid >> 5, lane = tid & 31;
    const int hi = lane >> 4, hi8 = hi * 8, col = lane & 15;
    const int pbase = blockIdx.x * 64 + wave * 16;
    __bf16* ps = psS[wave];
    __bf16* r1 = r1S[wave];

    const v8f kZero = (v8f)(0.f);

    // ps tile = [s1 | s2] per path, bf16 [16][128]
    {
        int row = lane >> 1, ch = lane & 1;
        int p = pbase + row;
        const float* base = (ch ? s2g : s1g) + (size_t)(p < n_paths ? p : 0) * 64;
        const float4* src = (const float4*)base;
        uint4* dst = (uint4*)(ps + row * 128 + ch * 64);
        float4 z = make_float4(0.f, 0.f, 0.f, 0.f);
#pragma unroll
        for (int q = 0; q < 8; ++q) {
            float4 f0 = (p < n_paths) ? src[2 * q]     : z;
            float4 f1 = (p < n_paths) ? src[2 * q + 1] : z;
            dst[q] = pack8(f0, f1);
        }
    }
    __builtin_amdgcn_wave_barrier();
    v16bf ap[4];
#pragma unroll
    for (int kt = 0; kt < 4; ++kt) ap[kt] = ldsA(ps, col, kt, hi, 16);

    // stage 1: selu(ps @ ro_w1 + b1) -> r1S (bf16)
    for (int nt = 0; nt < 16; ++nt) {
        v8f acc = wmma_bf16(ap[0], glbB(w1t, nt, 0, lane, 16), kZero);
#pragma unroll
        for (int kt = 1; kt < 4; ++kt)
            acc = wmma_bf16(ap[kt], glbB(w1t, nt, kt, lane, 16), acc);
        float b = rb1[nt * 16 + col];
#pragma unroll
        for (int v = 0; v < 8; ++v)
            r1[(v + hi8) * 256 + nt * 16 + col] = (__bf16)selu_f(acc[v] + b);
        __builtin_amdgcn_sched_barrier(0);
    }
    __builtin_amdgcn_wave_barrier();

    // stage 2: relu(r1 @ ro_w2 + b2), fused with the 256->1 dot against ro_w3
    float r3p[8];
#pragma unroll
    for (int v = 0; v < 8; ++v) r3p[v] = 0.f;
    for (int nt = 0; nt < 16; ++nt) {
        v8f acc = wmma_bf16(ldsA(r1, col, 0, hi, 32), glbB(w2t, nt, 0, lane, 32), kZero);
#pragma unroll
        for (int kt = 1; kt < 8; ++kt)
            acc = wmma_bf16(ldsA(r1, col, kt, hi, 32), glbB(w2t, nt, kt, lane, 32), acc);
        float b = rb2[nt * 16 + col];
        float w3 = rw3[nt * 16 + col];
#pragma unroll
        for (int v = 0; v < 8; ++v)
            r3p[v] += fmaxf(acc[v] + b, 0.f) * w3;
        __builtin_amdgcn_sched_barrier(0);
    }
    // reduce across the 16 lanes of each half-wave (wave32)
#pragma unroll
    for (int v = 0; v < 8; ++v) {
        for (int off = 8; off > 0; off >>= 1)
            r3p[v] += __shfl_xor(r3p[v], off, 16);
    }
    if (col == 0) {
#pragma unroll
        for (int v = 0; v < 8; ++v) r3S[wave][hi8 + v] = r3p[v] + rb3[0];
    }
    __builtin_amdgcn_wave_barrier();

    // final: o = fin_b + fin_w[0]*r + sum(ps * fin_w[1:129])
    if (hi == 0) {
        int p = pbase + col;
        if (p < n_paths) {
            float acc = finb[0] + finw[0] * r3S[wave][col];
            for (int i = 0; i < 128; ++i)
                acc += (float)ps[col * 128 + i] * finw[1 + i];
            out[p] = acc;
        }
    }
}

// ---------------------------------------------------------------- host launch

extern "C" void kernel_launch(void* const* d_in, const int* in_sizes, int n_in,
                              void* d_out, int out_size, void* d_ws, size_t ws_size,
                              hipStream_t stream) {
    const int*   links   = (const int*)  d_in[0];
    const int*   paths   = (const int*)  d_in[1];
    const int*   seqs    = (const int*)  d_in[2];
    const float* lfeat   = (const float*)d_in[3];
    const float* pfeat   = (const float*)d_in[4];
    const float* g1k     = (const float*)d_in[5];
    const float* g1rk    = (const float*)d_in[6];
    const float* g1b     = (const float*)d_in[7];
    const float* g2k     = (const float*)d_in[8];
    const float* g2rk    = (const float*)d_in[9];
    const float* g2b     = (const float*)d_in[10];
    const float* lu_w1   = (const float*)d_in[11];
    const float* lu_b1   = (const float*)d_in[12];
    const float* lu_w2   = (const float*)d_in[13];
    const float* lu_b2   = (const float*)d_in[14];
    const float* lu_w3   = (const float*)d_in[15];
    const float* lu_b3   = (const float*)d_in[16];
    const float* ro_w1   = (const float*)d_in[17];
    const float* ro_b1   = (const float*)d_in[18];
    const float* ro_w2   = (const float*)d_in[19];
    const float* ro_b2   = (const float*)d_in[20];
    const float* ro_w3   = (const float*)d_in[21];
    const float* ro_b3   = (const float*)d_in[22];
    const float* fin_w   = (const float*)d_in[23];
    const float* fin_b   = (const float*)d_in[24];

    const int E       = in_sizes[0];
    const int n_links = in_sizes[3] / 7;
    const int n_paths = in_sizes[4] / 11;

    char* w = (char*)d_ws;
    size_t off = 0;
    auto carve = [&](size_t bytes) -> void* {
        void* p = w + off;
        off += (bytes + 255) & ~(size_t)255;
        return p;
    };
    __bf16* ls   = (__bf16*)carve((size_t)n_links * 64 * sizeof(__bf16));
    float* agg   = (float*)carve((size_t)n_links * 64 * sizeof(float));
    float* s1    = (float*)carve((size_t)n_paths * 64 * sizeof(float));
    float* s2    = (float*)carve((size_t)n_paths * 64 * sizeof(float));
    int*   lens  = (int*)  carve((size_t)n_paths * sizeof(int));
    int*   rowst = (int*)  carve((size_t)n_paths * sizeof(int));
    __bf16* w1t  = (__bf16*)carve((size_t)256 * 128 * sizeof(__bf16));
    __bf16* w2t  = (__bf16*)carve((size_t)256 * 256 * sizeof(__bf16));
    (void)ws_size;

    hipMemsetAsync(lens, 0, (size_t)n_paths * sizeof(int), stream);
    k_init_links<<<(n_links * 64 + 255) / 256, 256, 0, stream>>>(lfeat, ls, n_links);
    k_init_paths<<<(n_paths * 64 + 255) / 256, 256, 0, stream>>>(pfeat, s1, s2, n_paths);
    k_rowinfo<<<(E + 255) / 256, 256, 0, stream>>>(paths, seqs, lens, rowst, E);
    k_prep_ro<<<(256 * 128 + 256 * 256 + 255) / 256, 256, 0, stream>>>(ro_w1, ro_w2, w1t, w2t);

    const int gruBlocks = (n_paths + 63) / 64;
    for (int it = 0; it < 8; ++it) {
        hipMemsetAsync(agg, 0, (size_t)n_links * 64 * sizeof(float), stream);
        k_gru<<<gruBlocks, 128, 0, stream>>>(links, rowst, lens, ls, agg, s1, s2,
                                             g1k, g1rk, g1b, g2k, g2rk, g2b, n_paths);
        k_linkupd<<<n_links, 128, 0, stream>>>(agg, ls, lu_w1, lu_b1, lu_w2, lu_b2, lu_w3, lu_b3);
    }

    k_readout<<<(n_paths + 63) / 64, 128, 0, stream>>>(
        s1, s2, (const uint4*)w1t, (const uint4*)w2t,
        ro_b1, ro_b2, ro_w3, ro_b3, fin_w, fin_b, (float*)d_out, n_paths);
}
